// FeedForwardApproximateBNN_43198781063321
// MI455X (gfx1250) — matrix operand order repaired and checked by reading, not compiled, not run on hardware
//
#include <hip/hip_runtime.h>

// ---------------------------------------------------------------------------
// FeedForwardApproximateBNN for MI455X (gfx1250, wave32, WMMA)
// 6x [GEMM (bf16 WMMA, fp32 accum) + bias + per-column activation]
// Async global->LDS (ASYNCcnt) double-buffered pipeline, 64x64 wave tiles,
// 64-wide staged K chunks, statically scheduled (unroll-by-2) buffers.
// ---------------------------------------------------------------------------

typedef __attribute__((ext_vector_type(16))) __bf16 v16bf;
typedef __attribute__((ext_vector_type(8)))  float  v8f;

#define BM    256   // block tile rows (batch)
#define BN    128   // block tile cols (output features)
#define CHK    64   // K halfs staged per LDS buffer (2 WMMA k-steps)
#define LSTR   72   // LDS row stride in halfs (64 data + 8 pad, 16B aligned)

union Frag { v16bf v; uint4 q[2]; };

__device__ __forceinline__ float apply_act(float h, int id) {
    if (id == 0) return h > 0.0f ? h : 0.0f;                 // relu
    if (id == 1) return 1.0f / (1.0f + __expf(-h));          // sigmoid
    if (id == 2) return tanhf(h);                            // tanh
    if (id == 3) return h >= 0.0f ? h : 0.1f * h;            // leaky relu
    const float scale = 1.0507009873554805f;                 // selu
    const float alpha = 1.6732632423543772f;
    return h > 0.0f ? scale * h : scale * alpha * (__expf(h) - 1.0f);
}

// Async 16B copy global -> LDS (GLOBAL_LOAD_ASYNC_TO_LDS_B128, ASYNCcnt).
__device__ __forceinline__ void async_copy_b128(const void* g, const void* lds) {
    const unsigned loff = (unsigned)(unsigned long long)(uintptr_t)lds;  // addr[31:0]
    const unsigned long long ga = (unsigned long long)(uintptr_t)g;
    asm volatile("global_load_async_to_lds_b128 %0, %1, off"
                 :: "v"(loff), "v"(ga) : "memory");
}

__device__ __forceinline__ void wait_async0() {
    asm volatile("s_wait_asynccnt 0x0" ::: "memory");
}

// Stage one 64-half K chunk: A rows (8 x b128 / thread), B rows (4 x b128).
__device__ __forceinline__ void stage_chunk(const __bf16* agRow, const __bf16* wgRow,
                                            __bf16* aL, __bf16* bL, int chunk) {
    const __bf16* ag = agRow + (size_t)chunk * CHK;
    const __bf16* wg = wgRow + (size_t)chunk * CHK;
#pragma unroll
    for (int c = 0; c < 8; ++c) async_copy_b128(ag + c * 8, aL + c * 8);
#pragma unroll
    for (int c = 0; c < 4; ++c) async_copy_b128(wg + c * 8, bL + c * 8);
}

// Two WMMA k-steps over one staged chunk. aB/bB are per-wave fragment bases.
__device__ __forceinline__ void compute_chunk(const __bf16* aB, const __bf16* bB,
                                              int h, v8f acc[4][4]) {
#pragma unroll
    for (int s = 0; s < 2; ++s) {
        // B frags (32x16): lane = column, 16 consecutive K halfs at 16h.
        Frag bF[4];
#pragma unroll
        for (int nt = 0; nt < 4; ++nt) {
            const __bf16* p = bB + nt * 16 * LSTR + s * 32 + h * 16;
            bF[nt].q[0] = *(const uint4*)(p);
            bF[nt].q[1] = *(const uint4*)(p + 8);
        }
        // A frags (16x32): lane half h holds K {16h..16h+15}.
#pragma unroll
        for (int mt = 0; mt < 4; ++mt) {
            Frag aF;
            const __bf16* p = aB + mt * 16 * LSTR + s * 32 + h * 16;
            aF.q[0] = *(const uint4*)(p);
            aF.q[1] = *(const uint4*)(p + 8);
#pragma unroll
            for (int nt = 0; nt < 4; ++nt)
                acc[mt][nt] = __builtin_amdgcn_wmma_f32_16x16x32_bf16(
                    false, aF.v, false, bF[nt].v,
                    (short)0, acc[mt][nt], false, false);
        }
    }
}

// fp32 -> bf16 (RNE), 4 elems/thread. n is a multiple of 4.
extern "C" __global__ void __launch_bounds__(256)
bnn_cvt_f32_bf16(const float* __restrict__ src, unsigned short* __restrict__ dst,
                 long long n) {
    long long i = ((long long)blockIdx.x * blockDim.x + threadIdx.x) * 4;
    if (i + 3 < n) {
        const float4 f = *(const float4*)(src + i);
        __bf16 b0 = (__bf16)f.x, b1 = (__bf16)f.y, b2 = (__bf16)f.z, b3 = (__bf16)f.w;
        ushort4 u;
        u.x = __builtin_bit_cast(unsigned short, b0);
        u.y = __builtin_bit_cast(unsigned short, b1);
        u.z = __builtin_bit_cast(unsigned short, b2);
        u.w = __builtin_bit_cast(unsigned short, b3);
        *(ushort4*)(dst + i) = u;
    }
}

// C[M,N] = act( A[M,K] x W[N,K]^T + bias[N] ), A/W bf16, accum fp32.
// 256 threads = 8 wave32 in 4(M) x 2(N) grid; each wave owns a 64x64 tile.
extern "C" __global__ void __launch_bounds__(256)
bnn_gemm_bias_act(const unsigned short* __restrict__ Aus,
                  const unsigned short* __restrict__ Wus,
                  const float* __restrict__ bias,
                  const int*   __restrict__ ids,
                  unsigned short* __restrict__ outBF,
                  float* __restrict__ outF32,
                  int N, int K, int storeF32) {
    const __bf16* A = (const __bf16*)Aus;
    const __bf16* W = (const __bf16*)Wus;

    __shared__ __align__(16) __bf16 Ash[2][BM * LSTR];   // 2 x 36 KB
    __shared__ __align__(16) __bf16 Bsh[2][BN * LSTR];   // 2 x 18 KB

    const int tid   = threadIdx.x;
    const int lane  = tid & 31;
    const int wave  = tid >> 5;
    const int waveM = wave >> 1;      // 0..3 -> 64 rows each
    const int waveN = wave & 1;       // 0..1 -> 64 cols each
    const int r     = lane & 15;      // row/col within 16x16 fragment
    const int h     = lane >> 4;      // lane half

    const int blockM = blockIdx.y * BM;
    const int blockN = blockIdx.x * BN;

    // Async staging ownership:
    //   A: thread t copies the full 64-half row t (8 x b128)
    //   B: thread t copies 32 halfs of row t/2 (4 x b128)
    const __bf16* agRow = A + (size_t)(blockM + tid) * K;
    const __bf16* wgRow = W + (size_t)(blockN + (tid >> 1)) * K + ((tid & 1) << 5);
    __bf16* aL0 = &Ash[0][tid * LSTR];
    __bf16* aL1 = &Ash[1][tid * LSTR];
    __bf16* bL0 = &Bsh[0][(tid >> 1) * LSTR + ((tid & 1) << 5)];
    __bf16* bL1 = &Bsh[1][(tid >> 1) * LSTR + ((tid & 1) << 5)];

    // Per-wave fragment bases.
    const __bf16* aB0 = &Ash[0][(waveM * 64 + r) * LSTR];
    const __bf16* aB1 = &Ash[1][(waveM * 64 + r) * LSTR];
    const __bf16* bB0 = &Bsh[0][(waveN * 64 + r) * LSTR];
    const __bf16* bB1 = &Bsh[1][(waveN * 64 + r) * LSTR];

    v8f acc[4][4];
#pragma unroll
    for (int mt = 0; mt < 4; ++mt)
#pragma unroll
        for (int nt = 0; nt < 4; ++nt)
            acc[mt][nt] = (v8f){0.f, 0.f, 0.f, 0.f, 0.f, 0.f, 0.f, 0.f};

    const int nkc = K / CHK;          // 32 or 64 (always even)

    stage_chunk(agRow, wgRow, aL0, bL0, 0);

    for (int kc = 0; kc < nkc; kc += 2) {
        wait_async0();        // this wave's LDS writes for chunk kc landed
        __syncthreads();      // all waves landed; prior reads of buf1 retired
        if (kc + 1 < nkc) {
            stage_chunk(agRow, wgRow, aL1, bL1, kc + 1);
            if (kc + 2 < nkc) {
                __builtin_prefetch(agRow + (size_t)(kc + 2) * CHK, 0, 1);
                __builtin_prefetch(wgRow + (size_t)(kc + 2) * CHK, 0, 1);
            }
        }
        compute_chunk(aB0, bB0, h, acc);

        wait_async0();
        __syncthreads();
        if (kc + 2 < nkc) {
            stage_chunk(agRow, wgRow, aL0, bL0, kc + 2);
            if (kc + 3 < nkc) {
                __builtin_prefetch(agRow + (size_t)(kc + 3) * CHK, 0, 1);
                __builtin_prefetch(wgRow + (size_t)(kc + 3) * CHK, 0, 1);
            }
        }
        compute_chunk(aB1, bB1, h, acc);
    }

    // Epilogue: C/D layout -> lane holds col r, rows j + 8h of each tile.
#pragma unroll
    for (int nt = 0; nt < 4; ++nt) {
        const int gn = blockN + waveN * 64 + nt * 16 + r;
        const float bv = bias[gn];
        const int   id = ids[gn];
#pragma unroll
        for (int mt = 0; mt < 4; ++mt) {
            const int gm0 = blockM + waveM * 64 + mt * 16 + 8 * h;
#pragma unroll
            for (int j = 0; j < 8; ++j) {
                const float y = apply_act(acc[mt][nt][j] + bv, id);
                const size_t o = (size_t)(gm0 + j) * N + gn;
                if (storeF32) {
                    outF32[o] = y;
                } else {
                    __bf16 yb = (__bf16)y;
                    outBF[o] = __builtin_bit_cast(unsigned short, yb);
                }
            }
        }
    }
}

// ---------------------------------------------------------------------------
extern "C" void kernel_launch(void* const* d_in, const int* in_sizes, int n_in,
                              void* d_out, int out_size, void* d_ws, size_t ws_size,
                              hipStream_t stream) {
    (void)in_sizes; (void)n_in; (void)out_size; (void)ws_size;

    const int M = 8192;
    const int Ns[6] = {4096, 4096, 4096, 4096, 4096, 2048};
    const int Ks[6] = {2048, 4096, 4096, 4096, 4096, 4096};

    const float* x = (const float*)d_in[0];
    const float* w[6]; const float* b[6]; const int* ids[6];
    for (int i = 0; i < 6; ++i) {
        w[i]   = (const float*)d_in[1 + 3 * i];
        b[i]   = (const float*)d_in[2 + 3 * i];
        ids[i] = (const int*)  d_in[3 + 3 * i];
    }

    // Carve workspace: 6 bf16 weight buffers + 2 bf16 activation buffers.
    char* ws = (char*)d_ws;
    size_t off = 0;
    auto carve = [&](size_t bytes) {
        void* p = ws + off;
        off = (off + bytes + 255) & ~(size_t)255;
        return p;
    };
    unsigned short* wbf[6];
    for (int i = 0; i < 6; ++i)
        wbf[i] = (unsigned short*)carve((size_t)Ns[i] * Ks[i] * 2);
    unsigned short* actA = (unsigned short*)carve((size_t)M * 4096 * 2);
    unsigned short* actB = (unsigned short*)carve((size_t)M * 4096 * 2);

    auto cvt = [&](const float* src, unsigned short* dst, long long n) {
        const long long threads = n / 4;
        const int blocks = (int)((threads + 255) / 256);
        bnn_cvt_f32_bf16<<<blocks, 256, 0, stream>>>(src, dst, n);
    };

    // One-time (per launch) fp32 -> bf16 conversions.
    cvt(x, actA, (long long)M * Ks[0]);
    for (int i = 0; i < 6; ++i)
        cvt(w[i], wbf[i], (long long)Ns[i] * Ks[i]);

    // Layer chain, ping-ponging bf16 activations; last layer -> fp32 d_out.
    unsigned short* cur = actA;
    unsigned short* nxt = actB;
    for (int i = 0; i < 6; ++i) {
        dim3 grid(Ns[i] / BN, M / BM);
        const int isLast = (i == 5) ? 1 : 0;
        bnn_gemm_bias_act<<<grid, 256, 0, stream>>>(
            cur, wbf[i], b[i], ids[i], nxt, (float*)d_out,
            Ns[i], Ks[i], isLast);
        unsigned short* t = cur; cur = nxt; nxt = t;
    }
}
